// ScaledDotProductAttention_43258910605393
// MI455X (gfx1250) — compile-verified, matrix operand order checked by reading, not run on hardware
//
#include <hip/hip_runtime.h>

typedef __attribute__((ext_vector_type(16))) _Float16 v16h;
typedef __attribute__((ext_vector_type(8)))  _Float16 v8h;
typedef __attribute__((ext_vector_type(8)))  float    v8f;
typedef __attribute__((ext_vector_type(4)))  float    f32x4;

#define MASK_VALUE (-1000000.0f)
#define HEAD_D     64
#define KBLK       64           // keys per iteration (two WMMA K-steps deep)
#define WAVES      4            // waves per workgroup
#define MWAVE      32           // query rows per wave (two 16-row WMMA tiles)
#define BLOCK_M    (WAVES*MWAVE)// 128 query rows per workgroup

// LDS row strides (in halves): multiples of 8 halves (16B) keep ds_load_b128
// aligned; stride 72 gives a 36-bank step -> conflict-free across 16 lanes.
#define K_STRIDE 72
#define V_STRIDE 72
#define P_STRIDE 72

static __device__ __forceinline__ v8h pack8(f32x4 a, f32x4 b, float s) {
  v8h r;
  r[0] = (_Float16)(a[0] * s); r[1] = (_Float16)(a[1] * s);
  r[2] = (_Float16)(a[2] * s); r[3] = (_Float16)(a[3] * s);
  r[4] = (_Float16)(b[0] * s); r[5] = (_Float16)(b[1] * s);
  r[6] = (_Float16)(b[2] * s); r[7] = (_Float16)(b[3] * s);
  return r;
}

static __device__ __forceinline__ v16h cat16(v8h lo, v8h hi) {
  return __builtin_shufflevector(lo, hi, 0,1,2,3,4,5,6,7,8,9,10,11,12,13,14,15);
}

// 16-half WMMA fragment from an LDS row: halves [g*8 .. +7] -> VGPRs 0-3,
// halves [16+g*8 .. +7] -> VGPRs 4-7 (CDNA5 16-bit A/B layout, g = lane/16).
static __device__ __forceinline__ v16h lds_frag(const _Float16* row, int g) {
  v8h lo = *(const v8h*)(row + g * 8);
  v8h hi = *(const v8h*)(row + 16 + g * 8);
  return cat16(lo, hi);
}

static __device__ __forceinline__ v8f wmma_f16(v16h a, v16h b, v8f c) {
  return __builtin_amdgcn_wmma_f32_16x16x32_f16(
      /*neg_a=*/false, a, /*neg_b=*/false, b,
      /*c_mod=*/(short)0, c, /*reuse_a=*/false, /*reuse_b=*/false);
}

// Online-softmax update for one 16-row tile over a 64-key block.
// Row m = v + 8*g of the C layout lives in one 16-lane half; exp'd
// probabilities are published (f16) to this wave's private P region.
static __device__ __forceinline__ void softmax_update(
    v8f st[4], v8f acc[4], float mrun[8], float lrun[8],
    _Float16* prow0, int n, int g) {
#pragma unroll
  for (int v = 0; v < 8; ++v) {
    float x = fmaxf(fmaxf(st[0][v], st[1][v]), fmaxf(st[2][v], st[3][v]));
    x = fmaxf(x, __shfl_xor(x, 1, 32));
    x = fmaxf(x, __shfl_xor(x, 2, 32));
    x = fmaxf(x, __shfl_xor(x, 4, 32));
    x = fmaxf(x, __shfl_xor(x, 8, 32));
    const float mnew  = fmaxf(mrun[v], x);
    const float alpha = __expf(mrun[v] - mnew);
    mrun[v] = mnew;
    const float e0 = __expf(st[0][v] - mnew);
    const float e1 = __expf(st[1][v] - mnew);
    const float e2 = __expf(st[2][v] - mnew);
    const float e3 = __expf(st[3][v] - mnew);
    float sm = (e0 + e1) + (e2 + e3);
    sm += __shfl_xor(sm, 1, 32);
    sm += __shfl_xor(sm, 2, 32);
    sm += __shfl_xor(sm, 4, 32);
    sm += __shfl_xor(sm, 8, 32);
    lrun[v] = lrun[v] * alpha + sm;
    acc[0][v] *= alpha; acc[1][v] *= alpha;
    acc[2][v] *= alpha; acc[3][v] *= alpha;
    _Float16* pr = prow0 + (v + 8 * g) * P_STRIDE;
    pr[n]      = (_Float16)e0;
    pr[16 + n] = (_Float16)e1;
    pr[32 + n] = (_Float16)e2;
    pr[48 + n] = (_Float16)e3;
  }
}

__global__ __launch_bounds__(128)
void sdpa_flash_f16wmma(const float* __restrict__ Q, const float* __restrict__ K,
                        const float* __restrict__ V, const int* __restrict__ valid_lens,
                        float* __restrict__ Out, int Sq, int Sk, int mblocks) {
  __shared__ _Float16 Klds[KBLK][K_STRIDE];       // [key][d]   (f16)
  __shared__ _Float16 Vlds[HEAD_D][V_STRIDE];     // [d][key]   (transposed f16)
  __shared__ _Float16 Plds[WAVES][32][P_STRIDE];  // per-wave exp'd scores [m][k]

  const int tid  = threadIdx.x;
  const int lane = tid & 31;
  const int w    = tid >> 5;        // wave id 0..3
  const int n    = lane & 15;       // column / row-in-tile index
  const int g    = lane >> 4;       // lane group (0 or 1)

  const int b  = blockIdx.x / mblocks;
  const int mb = blockIdx.x % mblocks;
  const int rowbase = mb * BLOCK_M + w * MWAVE;   // rows [rowbase, rowbase+32)

  const int vlen = valid_lens[b];

  const float* Qb = Q + (size_t)b * Sq * HEAD_D;
  const float* Kb = K + (size_t)b * Sk * HEAD_D;
  const float* Vb = V + (size_t)b * Sk * HEAD_D;

  // ---- Q A-fragments for two 16-row tiles, pre-scaled by 1/sqrt(d) ----
  const float scale = 0.125f;  // 1/sqrt(64)
  v16h qA0, qA1, qB0, qB1;
  {
    const float* qr = Qb + (size_t)(rowbase + n) * HEAD_D;
    f32x4 a0 = *(const f32x4*)(qr + g * 8);
    f32x4 a1 = *(const f32x4*)(qr + g * 8 + 4);
    f32x4 b0 = *(const f32x4*)(qr + 16 + g * 8);
    f32x4 b1 = *(const f32x4*)(qr + 16 + g * 8 + 4);
    qA0 = cat16(pack8(a0, a1, scale), pack8(b0, b1, scale));
    a0 = *(const f32x4*)(qr + 32 + g * 8);
    a1 = *(const f32x4*)(qr + 32 + g * 8 + 4);
    b0 = *(const f32x4*)(qr + 48 + g * 8);
    b1 = *(const f32x4*)(qr + 48 + g * 8 + 4);
    qA1 = cat16(pack8(a0, a1, scale), pack8(b0, b1, scale));
    qr += 16 * HEAD_D;  // second row tile
    a0 = *(const f32x4*)(qr + g * 8);
    a1 = *(const f32x4*)(qr + g * 8 + 4);
    b0 = *(const f32x4*)(qr + 16 + g * 8);
    b1 = *(const f32x4*)(qr + 16 + g * 8 + 4);
    qB0 = cat16(pack8(a0, a1, scale), pack8(b0, b1, scale));
    a0 = *(const f32x4*)(qr + 32 + g * 8);
    a1 = *(const f32x4*)(qr + 32 + g * 8 + 4);
    b0 = *(const f32x4*)(qr + 48 + g * 8);
    b1 = *(const f32x4*)(qr + 48 + g * 8 + 4);
    qB1 = cat16(pack8(a0, a1, scale), pack8(b0, b1, scale));
  }

  // ---- flash-attention state (two row tiles) ----
  v8f accA[4] = {}, accB[4] = {};
  float mrunA[8], lrunA[8], mrunB[8], lrunB[8];
#pragma unroll
  for (int v = 0; v < 8; ++v) {
    mrunA[v] = -__builtin_inff(); lrunA[v] = 0.0f;
    mrunB[v] = -__builtin_inff(); lrunB[v] = 0.0f;
  }

  // cooperative loader coords: 128 threads cover 64 rows x 2 col-groups of 32
  const int lr = tid >> 1;          // 0..63 key row
  const int lc = (tid & 1) << 5;    // 0 or 32 d column

  for (int kb = 0; kb < Sk; kb += KBLK) {
    __syncthreads();  // previous iteration's K/V reads complete

    // ---- stage K (64x64) and V^T (64x64) into LDS as f16 ----
    {
      const float* kg = Kb + (size_t)(kb + lr) * HEAD_D + lc;
      f32x4 k0 = *(const f32x4*)(kg);
      f32x4 k1 = *(const f32x4*)(kg + 4);
      f32x4 k2 = *(const f32x4*)(kg + 8);
      f32x4 k3 = *(const f32x4*)(kg + 12);
      f32x4 k4 = *(const f32x4*)(kg + 16);
      f32x4 k5 = *(const f32x4*)(kg + 20);
      f32x4 k6 = *(const f32x4*)(kg + 24);
      f32x4 k7 = *(const f32x4*)(kg + 28);
      *(v8h*)(&Klds[lr][0] + lc)      = pack8(k0, k1, 1.0f);
      *(v8h*)(&Klds[lr][0] + lc + 8)  = pack8(k2, k3, 1.0f);
      *(v8h*)(&Klds[lr][0] + lc + 16) = pack8(k4, k5, 1.0f);
      *(v8h*)(&Klds[lr][0] + lc + 24) = pack8(k6, k7, 1.0f);

      const float* vg = Vb + (size_t)(kb + lr) * HEAD_D + lc;
      f32x4 vq[8];
#pragma unroll
      for (int i = 0; i < 8; ++i) vq[i] = *(const f32x4*)(vg + 4 * i);
#pragma unroll
      for (int i = 0; i < 8; ++i) {
        Vlds[lc + 4 * i + 0][lr] = (_Float16)vq[i][0];
        Vlds[lc + 4 * i + 1][lr] = (_Float16)vq[i][1];
        Vlds[lc + 4 * i + 2][lr] = (_Float16)vq[i][2];
        Vlds[lc + 4 * i + 3][lr] = (_Float16)vq[i][3];
      }

      if (kb + KBLK < Sk) {  // uniform branch: prefetch next block into L2
        __builtin_prefetch(Kb + (size_t)(kb + KBLK + lr) * HEAD_D + lc, 0, 0);
        __builtin_prefetch(Vb + (size_t)(kb + KBLK + lr) * HEAD_D + lc, 0, 0);
      }
    }
    __syncthreads();

    // ---- S = (Q*scale) @ K^T : K fragments reused by both row tiles ----
    v8f sA[4], sB[4];
    {
      v8f z = {};
#pragma unroll
      for (int t = 0; t < 4; ++t) {
        v16h kf_lo = lds_frag(&Klds[t * 16 + n][0], g);
        v16h kf_hi = lds_frag(&Klds[t * 16 + n][32], g);
        sA[t] = wmma_f16(qA0, kf_lo, z);
        sA[t] = wmma_f16(qA1, kf_hi, sA[t]);
        sB[t] = wmma_f16(qB0, kf_lo, z);
        sB[t] = wmma_f16(qB1, kf_hi, sB[t]);
      }
    }

    // ---- mask (boundary / tail blocks only; uniform scalar branch) ----
    if (kb + KBLK > vlen) {
#pragma unroll
      for (int t = 0; t < 4; ++t) {
        const bool ok = (kb + t * 16 + n) < vlen;
#pragma unroll
        for (int v = 0; v < 8; ++v) {
          sA[t][v] = ok ? sA[t][v] : MASK_VALUE;
          sB[t][v] = ok ? sB[t][v] : MASK_VALUE;
        }
      }
    }

    // ---- online softmax + publish P (wave-private region) ----
    softmax_update(sA, accA, mrunA, lrunA, &Plds[w][0][0],  n, g);
    softmax_update(sB, accB, mrunB, lrunB, &Plds[w][16][0], n, g);

    // P is wave-private: LDS is in-order within a wave (DScnt), so only a
    // compiler-level ordering + wavefront fence is needed, not s_barrier.
    __builtin_amdgcn_fence(__ATOMIC_RELEASE, "wavefront");
    __builtin_amdgcn_wave_barrier();
    __builtin_amdgcn_fence(__ATOMIC_ACQUIRE, "wavefront");

    // ---- O += P @ V : V fragments reused by both row tiles ----
    {
      v16h pfA0 = lds_frag(&Plds[w][n][0],       g);  // tile A, keys 0..31
      v16h pfA1 = lds_frag(&Plds[w][n][32],      g);  // tile A, keys 32..63
      v16h pfB0 = lds_frag(&Plds[w][16 + n][0],  g);
      v16h pfB1 = lds_frag(&Plds[w][16 + n][32], g);
#pragma unroll
      for (int j = 0; j < 4; ++j) {
        v16h vf_lo = lds_frag(&Vlds[j * 16 + n][0],  g);
        v16h vf_hi = lds_frag(&Vlds[j * 16 + n][32], g);
        accA[j] = wmma_f16(pfA0, vf_lo, accA[j]);
        accA[j] = wmma_f16(pfA1, vf_hi, accA[j]);
        accB[j] = wmma_f16(pfB0, vf_lo, accB[j]);
        accB[j] = wmma_f16(pfB1, vf_hi, accB[j]);
      }
    }
  }

  // ---- epilogue: normalize and store (lanes 0-15 write 64B runs per (v,j)) ----
#pragma unroll
  for (int v = 0; v < 8; ++v) {
    const int rowA = rowbase + v + 8 * g;
    const float invA = 1.0f / lrunA[v];
    float* orowA = Out + ((size_t)b * Sq + rowA) * HEAD_D;
    orowA[n]      = accA[0][v] * invA;
    orowA[16 + n] = accA[1][v] * invA;
    orowA[32 + n] = accA[2][v] * invA;
    orowA[48 + n] = accA[3][v] * invA;

    const int rowB = rowA + 16;
    const float invB = 1.0f / lrunB[v];
    float* orowB = Out + ((size_t)b * Sq + rowB) * HEAD_D;
    orowB[n]      = accB[0][v] * invB;
    orowB[16 + n] = accB[1][v] * invB;
    orowB[32 + n] = accB[2][v] * invB;
    orowB[48 + n] = accB[3][v] * invB;
  }
}

extern "C" void kernel_launch(void* const* d_in, const int* in_sizes, int n_in,
                              void* d_out, int out_size, void* d_ws, size_t ws_size,
                              hipStream_t stream) {
  const float* Q  = (const float*)d_in[0];
  const float* K  = (const float*)d_in[1];
  const float* V  = (const float*)d_in[2];
  const int*   vl = (const int*)d_in[3];
  float* out = (float*)d_out;

  const int B  = in_sizes[3];                    // valid_lens: one entry per batch
  const int Sq = in_sizes[0] / (B * HEAD_D);     // 2048
  const int Sk = in_sizes[1] / (B * HEAD_D);     // 2048
  const int mblocks = Sq / BLOCK_M;              // 16

  dim3 grid(B * mblocks), block(WAVES * 32);
  hipLaunchKernelGGL(sdpa_flash_f16wmma, grid, block, 0, stream,
                     Q, K, V, vl, out, Sq, Sk, mblocks);
}
